// MultiheadAttention_31447750541850
// MI455X (gfx1250) — compile-verified, matrix operand order checked by reading, not run on hardware
//
#include <hip/hip_runtime.h>
#include <hip/hip_bf16.h>

// MI455X / gfx1250 fused multi-head attention, bf16 WMMA pipeline.
// cvt(W)->bf16 ; proj(q|k|v) -> bf16 ws ; flash attention -> bf16 ws ; proj -> f32 out.
// GEMM tile staging uses CDNA5 async global->LDS copies (ASYNCcnt) where the
// source is already bf16.

#define D_MODEL   1024
#define NUM_HEADS 16
#define HEAD_DIM  64
#define NBATCH    8
#define SEQ       1024

typedef __attribute__((ext_vector_type(16))) __bf16 bf16x16;
typedef __attribute__((ext_vector_type(8)))  float  f32x8;

union FragU { bf16x16 v; uint4 u[2]; };

__device__ __forceinline__ unsigned short f32_to_bf16_bits(float f) {
    unsigned int u = __float_as_uint(f);
    u += 0x7FFFu + ((u >> 16) & 1u);           // round-to-nearest-even
    return (unsigned short)(u >> 16);
}

__device__ __forceinline__ unsigned pack_bf16x2(float lo, float hi) {
    return (unsigned)f32_to_bf16_bits(lo) | ((unsigned)f32_to_bf16_bits(hi) << 16);
}

// Generic LDS pointer -> 32-bit LDS byte address (aperture truncation, ISA 10.2).
__device__ __forceinline__ unsigned lds_addr_of(const void* p) {
    return (unsigned)(unsigned long long)p;
}

// CDNA5 async bulk copy: per-lane 16B global -> LDS, tracked by ASYNCcnt.
__device__ __forceinline__ void async_copy_b128(unsigned lds, const unsigned short* g) {
    asm volatile("global_load_async_to_lds_b128 %0, %1, off"
                 :: "v"(lds), "v"(g) : "memory");
}
__device__ __forceinline__ void wait_async0() {
    asm volatile("s_wait_asynccnt 0x0" ::: "memory");
}

// Load a 16x32 bf16 fragment (A-layout / row-per-lane B-layout) from a
// row-major bf16 matrix. Lane L: row = L%16, half = L/16.
// VGPRs 0-3 hold K = half*8 + 0..7 (contiguous 16B), VGPRs 4-7 hold +16.
__device__ __forceinline__ FragU load_frag(const unsigned short* p, int stride) {
    const int lane = threadIdx.x & 31;
    const int m    = lane & 15;
    const int half = lane >> 4;
    FragU f;
    f.u[0] = *(const uint4*)(p + (size_t)m * stride + half * 8);
    f.u[1] = *(const uint4*)(p + (size_t)m * stride + 16 + half * 8);
    return f;
}

__device__ __forceinline__ f32x8 wmma_bf16(const FragU& a, const FragU& b, f32x8 c) {
    return __builtin_amdgcn_wmma_f32_16x16x32_bf16(
        /*neg_a=*/false, a.v, /*neg_b=*/false, b.v,
        /*c_mod=*/(short)0, c, /*reuse_a=*/false, /*reuse_b=*/false);
}

// ---------------------------------------------------------------------------
// One-shot f32 -> bf16 conversion (vectorized 8 elements/thread).
// ---------------------------------------------------------------------------
__global__ __launch_bounds__(256) void cvt_f32_bf16_kernel(
    const float* __restrict__ src, unsigned short* __restrict__ dst) {
    const size_t i = ((size_t)blockIdx.x * 256 + threadIdx.x) * 8;
    const float4 a = *(const float4*)(src + i);
    const float4 b = *(const float4*)(src + i + 4);
    uint4 o;
    o.x = pack_bf16x2(a.x, a.y);
    o.y = pack_bf16x2(a.z, a.w);
    o.z = pack_bf16x2(b.x, b.y);
    o.w = pack_bf16x2(b.z, b.w);
    *(uint4*)(dst + i) = o;
}

// ---------------------------------------------------------------------------
// C[M=8192][1024] = A[M][1024] @ W^T + b.  Block tile 128x64, 8 waves,
// each wave a 2x2 grid of 16x16 WMMA tiles, K staged 64-wide in LDS
// (two 32-deep WMMA sub-steps per barrier round). W arrives pre-converted
// to bf16; bf16 tiles are staged with async global->LDS copies.
// ---------------------------------------------------------------------------
template <bool IN_BF16, bool OUT_BF16>
__global__ __launch_bounds__(256) void proj_gemm_kernel(
    const void* __restrict__ Ain, const unsigned short* __restrict__ Wbf,
    const float* __restrict__ bias, void* __restrict__ Cout) {
    __shared__ unsigned short As[128][64];   // 16 KB
    __shared__ unsigned short Bs[64][64];    // 8 KB

    const int tid  = threadIdx.x;
    const int m0   = blockIdx.x * 128;
    const int n0   = blockIdx.y * 64;
    const int w    = tid >> 5;
    const int wm   = w >> 1;          // 0..3
    const int wn   = w & 1;           // 0..1
    const int lane = tid & 31;

    const f32x8 zero = {0.f,0.f,0.f,0.f,0.f,0.f,0.f,0.f};
    f32x8 acc[2][2] = {{zero, zero}, {zero, zero}};

    for (int k0 = 0; k0 < D_MODEL; k0 += 64) {
        // ---- stage B tile = bf16 W rows (64x64), async ----
        {
            const int r  = tid >> 2;
            const int cs = (tid & 3) * 16;
            const unsigned short* src = Wbf + (size_t)(n0 + r) * D_MODEL + k0 + cs;
            async_copy_b128(lds_addr_of(&Bs[r][cs]),     src);
            async_copy_b128(lds_addr_of(&Bs[r][cs + 8]), src + 8);
        }
        // ---- stage A tile (128x64) ----
        {
            const int r  = tid >> 1;
            const int cs = (tid & 1) * 32;
            if (IN_BF16) {
                const unsigned short* src = (const unsigned short*)Ain +
                                            (size_t)(m0 + r) * D_MODEL + k0 + cs;
#pragma unroll
                for (int j = 0; j < 4; ++j)
                    async_copy_b128(lds_addr_of(&As[r][cs + 8 * j]), src + 8 * j);
            } else {
                const float* src = (const float*)Ain + (size_t)(m0 + r) * D_MODEL + k0 + cs;
#pragma unroll
                for (int j = 0; j < 32; ++j) As[r][cs + j] = f32_to_bf16_bits(src[j]);
            }
        }
        wait_async0();           // per-wave: our async writes have landed in LDS
        __syncthreads();         // tile visible to all waves

#pragma unroll
        for (int ks = 0; ks < 2; ++ks) {
            FragU a0 = load_frag(&As[wm * 32][ks * 32], 64);
            FragU a1 = load_frag(&As[wm * 32 + 16][ks * 32], 64);
            FragU b0 = load_frag(&Bs[wn * 32][ks * 32], 64);
            FragU b1 = load_frag(&Bs[wn * 32 + 16][ks * 32], 64);

            acc[0][0] = wmma_bf16(a0, b0, acc[0][0]);
            acc[0][1] = wmma_bf16(a0, b1, acc[0][1]);
            acc[1][0] = wmma_bf16(a1, b0, acc[1][0]);
            acc[1][1] = wmma_bf16(a1, b1, acc[1][1]);
        }
        __syncthreads();         // all reads done before next round's staging
    }

    // epilogue: C layout — lane L holds (row r + 8*(L/16), col L%16) in VGPR r
    const int g  = lane >> 4;
    const int nc = lane & 15;
#pragma unroll
    for (int mt = 0; mt < 2; ++mt) {
#pragma unroll
        for (int nt = 0; nt < 2; ++nt) {
            const int   gcol = n0 + wn * 32 + nt * 16 + nc;
            const float bv   = bias[gcol];
#pragma unroll
            for (int r = 0; r < 8; ++r) {
                const int   grow = m0 + wm * 32 + mt * 16 + r + 8 * g;
                const float val  = acc[mt][nt][r] + bv;
                if (OUT_BF16)
                    ((unsigned short*)Cout)[(size_t)grow * D_MODEL + gcol] = f32_to_bf16_bits(val);
                else
                    ((float*)Cout)[(size_t)grow * D_MODEL + gcol] = val;
            }
        }
    }
}

// ---------------------------------------------------------------------------
// Flash-style attention. Block = 4 waves; one block per (batch, head, 64 q rows).
// Each wave owns 16 q rows; keys processed 32 at a time.
// Output written in the reference's (h,n,s,d)-transposed flat layout (bf16).
// ---------------------------------------------------------------------------
__global__ __launch_bounds__(128) void flash_attn_kernel(
    const unsigned short* __restrict__ qp, const unsigned short* __restrict__ kp,
    const unsigned short* __restrict__ vp, const int* __restrict__ mask,
    unsigned short* __restrict__ attn_out) {
    __shared__ unsigned short Vts[64][32];      // V^T tile: [d][kpos], 4 KB
    __shared__ unsigned short Ps[4][16][32];    // per-wave P tile, 4 KB

    const int tid  = threadIdx.x;
    const int w    = tid >> 5;
    const int lane = tid & 31;
    const int g    = lane >> 4;
    const int nc   = lane & 15;
    const int q0   = blockIdx.x * 64;
    const int h    = blockIdx.y;
    const int nb   = blockIdx.z;

    // Q fragments straight from global (row-major [s][1024] matches A layout)
    const unsigned short* qbase =
        qp + ((size_t)nb * SEQ + q0 + w * 16) * D_MODEL + h * HEAD_DIM;
    const FragU qa0 = load_frag(qbase, D_MODEL);        // d 0..31
    const FragU qa1 = load_frag(qbase + 32, D_MODEL);   // d 32..63

    const f32x8 zero = {0.f,0.f,0.f,0.f,0.f,0.f,0.f,0.f};
    f32x8 O[4] = {zero, zero, zero, zero};
    float rm[8], rl[8];
#pragma unroll
    for (int r = 0; r < 8; ++r) { rm[r] = -3.0e38f; rl[r] = 0.f; }

    const float scale = 0.125f;  // 1/sqrt(HEAD_DIM)

    for (int kb = 0; kb < SEQ; kb += 32) {
        // ---- stage V^T tile (32 keys x 64 dims -> Vts[d][k]) ----
        {
            const int krow = tid >> 2;           // 0..31
            const int ds   = (tid & 3) * 16;     // 0,16,32,48
            const unsigned short* vrow =
                vp + ((size_t)nb * SEQ + kb + krow) * D_MODEL + h * HEAD_DIM + ds;
            union { uint4 q[2]; unsigned short s[16]; } u;
            u.q[0] = *(const uint4*)(vrow);
            u.q[1] = *(const uint4*)(vrow + 8);
#pragma unroll
            for (int j = 0; j < 16; ++j) Vts[ds + j][krow] = u.s[j];
        }
        __syncthreads();

        // ---- S = Q * K^T for 16 q rows x 32 keys ----
        const unsigned short* kbase =
            kp + ((size_t)nb * SEQ + kb) * D_MODEL + h * HEAD_DIM;
        const FragU k00 = load_frag(kbase, D_MODEL);                  // keys +0..15,  d 0..31
        const FragU k01 = load_frag(kbase + 32, D_MODEL);             // keys +0..15,  d 32..63
        const FragU k10 = load_frag(kbase + 16 * D_MODEL, D_MODEL);   // keys +16..31, d 0..31
        const FragU k11 = load_frag(kbase + 16 * D_MODEL + 32, D_MODEL);

        f32x8 c0 = zero, c1 = zero;
        c0 = wmma_bf16(qa0, k00, c0);
        c0 = wmma_bf16(qa1, k01, c0);
        c1 = wmma_bf16(qa0, k10, c1);
        c1 = wmma_bf16(qa1, k11, c1);

        // ---- mask + online softmax (per-row reductions across 16-lane groups) ----
        float p0v[8], p1v[8], alpha[8];
#pragma unroll
        for (int r = 0; r < 8; ++r) {
            const int  qrow = q0 + w * 16 + r + 8 * g;
            const int* mrow = mask + ((size_t)nb * SEQ + qrow) * SEQ + kb;
            float s0 = (mrow[nc]      == 0) ? -1.0e9f : c0[r] * scale;
            float s1 = (mrow[16 + nc] == 0) ? -1.0e9f : c1[r] * scale;

            float v = fmaxf(s0, s1);
#pragma unroll
            for (int off = 8; off >= 1; off >>= 1) v = fmaxf(v, __shfl_xor(v, off, 32));
            const float mnew = fmaxf(rm[r], v);
            const float a    = __expf(rm[r] - mnew);
            const float p0   = __expf(s0 - mnew);
            const float p1   = __expf(s1 - mnew);
            float srow = p0 + p1;
#pragma unroll
            for (int off = 8; off >= 1; off >>= 1) srow += __shfl_xor(srow, off, 32);
            rl[r]    = rl[r] * a + srow;
            rm[r]    = mnew;
            alpha[r] = a;
            p0v[r]   = p0;
            p1v[r]   = p1;
        }
#pragma unroll
        for (int t = 0; t < 4; ++t)
#pragma unroll
            for (int r = 0; r < 8; ++r) O[t][r] *= alpha[r];

        // ---- re-layout P (C layout -> A layout) via per-wave LDS ----
#pragma unroll
        for (int r = 0; r < 8; ++r) {
            Ps[w][r + 8 * g][nc]      = f32_to_bf16_bits(p0v[r]);
            Ps[w][r + 8 * g][16 + nc] = f32_to_bf16_bits(p1v[r]);
        }
        const FragU pf = load_frag(&Ps[w][0][0], 32);   // wave-local; DS in-order

        // ---- O += P @ V ----
#pragma unroll
        for (int t = 0; t < 4; ++t) {
            const FragU vf = load_frag(&Vts[t * 16][0], 32);
            O[t] = wmma_bf16(pf, vf, O[t]);
        }
        __syncthreads();   // before Vts is overwritten next iteration
    }

    // ---- normalize and store in reference's transposed-flat layout ----
#pragma unroll
    for (int t = 0; t < 4; ++t) {
#pragma unroll
        for (int r = 0; r < 8; ++r) {
            const int   srow = q0 + w * 16 + r + 8 * g;
            const float val  = O[t][r] / rl[r];
            const size_t off =
                ((size_t)(h * 8 + nb) * SEQ + srow) * HEAD_DIM + t * 16 + nc;
            attn_out[off] = f32_to_bf16_bits(val);
        }
    }
}

// ---------------------------------------------------------------------------
extern "C" void kernel_launch(void* const* d_in, const int* in_sizes, int n_in,
                              void* d_out, int out_size, void* d_ws, size_t ws_size,
                              hipStream_t stream) {
    (void)in_sizes; (void)n_in; (void)out_size; (void)ws_size;

    const float* q    = (const float*)d_in[0];
    const float* k    = (const float*)d_in[1];
    const float* v    = (const float*)d_in[2];
    const int*   mask = (const int*)d_in[3];
    const float* W    = (const float*)d_in[4];
    const float* b    = (const float*)d_in[5];
    float*       out  = (float*)d_out;

    const size_t TEN = (size_t)NBATCH * SEQ * D_MODEL;   // 8M elements
    unsigned short* ws  = (unsigned short*)d_ws;
    unsigned short* qp  = ws;                // projected Q (bf16)
    unsigned short* kp  = ws + TEN;          // projected K
    unsigned short* vp  = ws + 2 * TEN;      // projected V
    unsigned short* ap  = ws + 3 * TEN;      // attention output (scrambled layout)
    unsigned short* wbf = ws + 4 * TEN;      // W in bf16 (1M elements)

    // W -> bf16 once (1M elements, 8/thread)
    cvt_f32_bf16_kernel<<<(D_MODEL * D_MODEL) / (256 * 8), 256, 0, stream>>>(W, wbf);

    const dim3 gemm_grid(NBATCH * SEQ / 128, D_MODEL / 64);   // (64, 16)
    proj_gemm_kernel<false, true><<<gemm_grid, 256, 0, stream>>>(q, wbf, b, qp);
    proj_gemm_kernel<false, true><<<gemm_grid, 256, 0, stream>>>(k, wbf, b, kp);
    proj_gemm_kernel<false, true><<<gemm_grid, 256, 0, stream>>>(v, wbf, b, vp);

    flash_attn_kernel<<<dim3(SEQ / 64, NUM_HEADS, NBATCH), 128, 0, stream>>>(
        qp, kp, vp, mask, ap);

    proj_gemm_kernel<true, false><<<gemm_grid, 256, 0, stream>>>(ap, wbf, b, out);
}
